// LSTMDecoder_83313775608351
// MI455X (gfx1250) — compile-verified, matrix operand order checked by reading, not run on hardware
//
#include <hip/hip_runtime.h>

// ---------------------------------------------------------------------------
// LSTM decoder for MI455X (gfx1250, wave32, WMMA).
// L=2, B=4096, H=1024, F=3, T=64.
//
// 3 GEMMs of [4096x1024]x[1024x4096] per step => 6.6 TFLOP total. bf16
// weights (24MB) + state (~48MB) are L2-resident (192MB); matrix-core bound
// => v_wmma_f32_16x16x32_bf16 with fp32 accumulation.
//
// Tile: 32 batch rows x 128 cols x 4 gates per 256-thread WG. Each wave owns
// 16 cols for all four gates and both row tiles (8 accumulators), so B
// fragments are reused 2x and the i/f/g/o pointwise update is wave-local.
// A tiles staged to LDS with CDNA5 async global->LDS loads, double-buffered
// so the layer-1 kernel overlaps its second A-tile transfer with GEMM-1.
// ---------------------------------------------------------------------------

typedef __bf16 v16bf __attribute__((ext_vector_type(16)));
typedef float  v8f   __attribute__((ext_vector_type(8)));
typedef unsigned int u32x4 __attribute__((ext_vector_type(4)));

union FragU {            // 32-byte WMMA 16-bit fragment (16 bf16 per lane)
    u32x4 q[2];
    v16bf v;
};

#define HDIM 1024
#define BDIM 4096
#define TDIM 64
#define FDIM 3
#define BM   32                  // batch rows per workgroup
#define KTILES (HDIM / 32)       // 32 K-iterations of depth 32
#define LDS_STRIDE 1032          // bf16 elems per row (+8 pad -> low-conflict b128)
#define LDS_BYTES (BM * LDS_STRIDE * 2)
#define GATE_STRIDE_ELEMS (64 * KTILES * 512)   // 2MB per gate in packed buffer

// ---------------------------------------------------------------------------
// Pack fp32 weight matrix W (N=4096 x K=1024, row-major, used as B = W^T) into
// per-lane bf16 WMMA B fragments:
//   P[((n_tile*KTILES + k_tile)*32 + lane)*16 .. +15]
// Assumed 16-bit B layout (mirrors documented A layout):
//   lane<16 : N = n_tile*16+lane,    K chunks {0..7} then {16..23}
//   lane>=16: N = n_tile*16+lane-16, K chunks {8..15} then {24..31}
// ---------------------------------------------------------------------------
__global__ void pack_weight_kernel(const float* __restrict__ W,
                                   __bf16* __restrict__ P) {
    int t    = blockIdx.x * blockDim.x + threadIdx.x; // one thread per fragment-lane
    int lane = t & 31;
    int kt   = (t >> 5) & (KTILES - 1);
    int nt   = t >> 10;                               // 0..255
    int n     = nt * 16 + (lane & 15);
    int kbase = kt * 32 + ((lane < 16) ? 0 : 8);

    FragU f;
    #pragma unroll
    for (int j = 0; j < 8; ++j)
        f.v[j] = (__bf16)W[(size_t)n * HDIM + kbase + j];
    #pragma unroll
    for (int j = 0; j < 8; ++j)
        f.v[8 + j] = (__bf16)W[(size_t)n * HDIM + kbase + 16 + j];

    u32x4* dst = (u32x4*)(P + ((size_t)(nt * KTILES + kt) * 32 + lane) * 16);
    dst[0] = f.q[0];
    dst[1] = f.q[1];
}

// Convert encoder state into workspace (bf16 h, fp32 c working copies).
__global__ void init_state_kernel(const float* __restrict__ eh,
                                  const float* __restrict__ ec,
                                  __bf16* __restrict__ h0, __bf16* __restrict__ h1,
                                  float* __restrict__ c0, float* __restrict__ c1,
                                  int n) {
    int i = blockIdx.x * blockDim.x + threadIdx.x;
    if (i < n) {
        h0[i] = (__bf16)eh[i];
        h1[i] = (__bf16)eh[n + i];
        c0[i] = ec[i];
        c1[i] = ec[n + i];
    }
}

__device__ __forceinline__ float sigm(float x) {
    return 1.0f / (1.0f + __expf(-x));
}

// ---------------------------------------------------------------------------
// One LSTM layer for one time step.
//   grid = (B/32, H/128), block = 256 (8 waves),
//   dynamic LDS = LDS_BYTES * (TWO_GEMM ? 2 : 1).
// TWO_GEMM: layer 1 accumulates A1@pW1^T + A2@pW2^T (h0_new and h1_prev);
//           both A tiles are async-staged up front into separate buffers.
// HAS_X   : layer 0 adds the K=3 input projection x@W_ih0^T in the epilogue.
// ---------------------------------------------------------------------------
template <bool TWO_GEMM, bool HAS_X>
__global__ __launch_bounds__(256)
void lstm_layer_kernel(const __bf16* __restrict__ A1,
                       const __bf16* __restrict__ pW1,
                       const __bf16* __restrict__ A2,
                       const __bf16* __restrict__ pW2,
                       __bf16* __restrict__ h_out,     // (B,H) bf16 in/out rows
                       float*  __restrict__ c_state,   // (B,H) fp32 in/out
                       const float* __restrict__ b_ih,
                       const float* __restrict__ b_hh,
                       const float* __restrict__ x, int x_stride, // (B,3)
                       const float* __restrict__ W_ih0)           // (4H,3)
{
    extern __shared__ __align__(16) __bf16 Asm[];   // 1 or 2 x (BM x LDS_STRIDE)

    const int tid   = threadIdx.x;
    const int lane  = tid & 31;
    const int wave  = tid >> 5;       // 0..7 -> column subtile
    const int btile = blockIdx.x;     // 0..B/BM-1
    const int ctile = blockIdx.y;     // 0..7

    v8f acc[2][4] = {};               // [row tile][gate i,f,g,o] 16x16 tiles

    const int m    = lane & 15;
    const int koff = (lane < 16) ? 0 : 8;

    // ---- async global -> LDS staging of one A tile (BM x 1024 bf16) ----
    // 16 async b128 ops per lane; ASYNCcnt tracks them, in-order completion.
    auto stage = [&](const __bf16* __restrict__ src, int buf) {
        int r  = tid >> 3;            // 0..31
        int c0 = (tid & 7) * 128;     // 128-element (256B) chunk per thread
        const char* g = (const char*)(src + (size_t)(btile * BM + r) * HDIM + c0);
        unsigned l = (unsigned)(uintptr_t)&Asm[buf * BM * LDS_STRIDE + r * LDS_STRIDE + c0];
        #pragma unroll
        for (int i = 0; i < 16; ++i) {
            asm volatile("global_load_async_to_lds_b128 %0, %1, off"
                         :: "v"(l + 16u * i), "v"(g + 16 * i) : "memory");
        }
    };

    auto run_gemm = [&](const __bf16* __restrict__ pW, int buf) {
        const __bf16* base = &Asm[buf * BM * LDS_STRIDE];
        const size_t wtile = (size_t)ctile * 8 + wave;    // g=0 n_tile index
        for (int kt = 0; kt < KTILES; ++kt) {
            FragU a0, a1;             // A fragments for the two 16-row tiles
            const __bf16* ar0 = base + m * LDS_STRIDE + kt * 32 + koff;
            const __bf16* ar1 = ar0 + 16 * LDS_STRIDE;
            a0.q[0] = *(const u32x4*)(ar0);
            a0.q[1] = *(const u32x4*)(ar0 + 16);
            a1.q[0] = *(const u32x4*)(ar1);
            a1.q[1] = *(const u32x4*)(ar1 + 16);
            // g=0 fragment pointer; gate streams are +2MB apart. Prefetch all
            // four streams 2 k_tiles ahead off this ONE base register with
            // immediate offsets (scope 0 = WGP => pulls into near caches).
            const __bf16* bp0 = pW + ((wtile * KTILES + kt) * 32 + lane) * 16;
            asm volatile(
                "global_prefetch_b8 %0, off offset:2048\n\t"
                "global_prefetch_b8 %0, off offset:2099200\n\t"
                "global_prefetch_b8 %0, off offset:4196352\n\t"
                "global_prefetch_b8 %0, off offset:6293504"
                :: "v"(bp0));
            #pragma unroll
            for (int g = 0; g < 4; ++g) {
                const __bf16* bp = bp0 + (size_t)g * GATE_STRIDE_ELEMS;
                FragU b;
                b.q[0] = ((const u32x4*)bp)[0];
                b.q[1] = ((const u32x4*)bp)[1];
                acc[0][g] = __builtin_amdgcn_wmma_f32_16x16x32_bf16(
                    false, a0.v, false, b.v, (short)0, acc[0][g], false, false);
                acc[1][g] = __builtin_amdgcn_wmma_f32_16x16x32_bf16(
                    false, a1.v, false, b.v, (short)0, acc[1][g], false, false);
            }
        }
    };

    if constexpr (TWO_GEMM) {
        stage(A1, 0);
        stage(A2, 1);                 // overlaps with GEMM-1 below
        asm volatile("s_wait_asynccnt 0x10" ::: "memory");  // buf0's 16 ops done
        __syncthreads();
        run_gemm(pW1, 0);
        asm volatile("s_wait_asynccnt 0x0" ::: "memory");   // buf1 done
        __syncthreads();
        run_gemm(pW2, 1);
    } else {
        stage(A1, 0);
        asm volatile("s_wait_asynccnt 0x0" ::: "memory");
        __syncthreads();
        run_gemm(pW1, 0);
    }

    // ---- epilogue: biases (+ x@W_ih0^T), gate nonlinearities, c/h update ----
    const int j = ctile * 128 + wave * 16 + (lane & 15); // column within H
    float bias[4];
    #pragma unroll
    for (int g = 0; g < 4; ++g)
        bias[g] = b_ih[g * HDIM + j] + b_hh[g * HDIM + j];

    float xw0[4], xw1[4], xw2[4];
    if constexpr (HAS_X) {
        #pragma unroll
        for (int g = 0; g < 4; ++g) {
            const float* wr = W_ih0 + (size_t)(g * HDIM + j) * FDIM;
            xw0[g] = wr[0]; xw1[g] = wr[1]; xw2[g] = wr[2];
        }
    }

    #pragma unroll
    for (int rt = 0; rt < 2; ++rt) {
        const int rbase = btile * BM + rt * 16 + ((lane < 16) ? 0 : 8);
        #pragma unroll
        for (int v = 0; v < 8; ++v) {
            int b = rbase + v;       // C/D layout: row = v (+8 for hi lanes)
            float gi = acc[rt][0][v] + bias[0];
            float gf = acc[rt][1][v] + bias[1];
            float gg = acc[rt][2][v] + bias[2];
            float go = acc[rt][3][v] + bias[3];
            if constexpr (HAS_X) {
                float x0 = x[(size_t)b * x_stride + 0];
                float x1 = x[(size_t)b * x_stride + 1];
                float x2 = x[(size_t)b * x_stride + 2];
                gi += x0 * xw0[0] + x1 * xw1[0] + x2 * xw2[0];
                gf += x0 * xw0[1] + x1 * xw1[1] + x2 * xw2[1];
                gg += x0 * xw0[2] + x1 * xw1[2] + x2 * xw2[2];
                go += x0 * xw0[3] + x1 * xw1[3] + x2 * xw2[3];
            }
            size_t idx  = (size_t)b * HDIM + j;
            float c_new = sigm(gf) * c_state[idx] + sigm(gi) * tanhf(gg);
            float h_new = sigm(go) * tanhf(c_new);
            c_state[idx] = c_new;
            h_out[idx]   = (__bf16)h_new;
        }
    }
}

// ---------------------------------------------------------------------------
// y = h1 @ W_fc^T + b_fc  (F=3). One wave per batch row; lane-strided dot +
// wave shuffle reduction. Writes d_out[b, t, :] (B, T, F).
// ---------------------------------------------------------------------------
__global__ void fc_out_kernel(const __bf16* __restrict__ h1,
                              const float* __restrict__ Wfc,
                              const float* __restrict__ bfc,
                              float* __restrict__ out, int t) {
    int wave = threadIdx.x >> 5;
    int lane = threadIdx.x & 31;
    int b    = blockIdx.x * 4 + wave;
    float s0 = 0.f, s1 = 0.f, s2 = 0.f;
    for (int k = lane; k < HDIM; k += 32) {
        float hv = (float)h1[(size_t)b * HDIM + k];
        s0 += hv * Wfc[k];
        s1 += hv * Wfc[HDIM + k];
        s2 += hv * Wfc[2 * HDIM + k];
    }
    #pragma unroll
    for (int off = 16; off > 0; off >>= 1) {
        s0 += __shfl_down(s0, off, 32);
        s1 += __shfl_down(s1, off, 32);
        s2 += __shfl_down(s2, off, 32);
    }
    if (lane == 0) {
        float* o = out + ((size_t)b * TDIM + t) * FDIM;
        o[0] = s0 + bfc[0];
        o[1] = s1 + bfc[1];
        o[2] = s2 + bfc[2];
    }
}

// ---------------------------------------------------------------------------
extern "C" void kernel_launch(void* const* d_in, const int* in_sizes, int n_in,
                              void* d_out, int out_size, void* d_ws, size_t ws_size,
                              hipStream_t stream) {
    const float* eh    = (const float*)d_in[0];   // (2,B,H)
    const float* ec    = (const float*)d_in[1];   // (2,B,H)
    const float* x0    = (const float*)d_in[2];   // (B,1,3)
    const float* Wih0  = (const float*)d_in[3];   // (4H,3)
    const float* Whh0  = (const float*)d_in[4];   // (4H,H)
    const float* bih0  = (const float*)d_in[5];
    const float* bhh0  = (const float*)d_in[6];
    const float* Wih1  = (const float*)d_in[7];   // (4H,H)
    const float* Whh1  = (const float*)d_in[8];   // (4H,H)
    const float* bih1  = (const float*)d_in[9];
    const float* bhh1  = (const float*)d_in[10];
    const float* Wfc   = (const float*)d_in[11];  // (3,H)
    const float* bfc   = (const float*)d_in[12];
    float* outF = (float*)d_out;                  // (B,T,3)

    // workspace layout (all 256B aligned by construction)
    char*  ws   = (char*)d_ws;
    size_t WPK  = (size_t)4 * HDIM * HDIM * sizeof(__bf16);   // 8MB per matrix
    size_t NBH  = (size_t)BDIM * HDIM;
    __bf16* pw_hh0 = (__bf16*)(ws);
    __bf16* pw_ih1 = (__bf16*)(ws + WPK);
    __bf16* pw_hh1 = (__bf16*)(ws + 2 * WPK);
    __bf16* h0     = (__bf16*)(ws + 3 * WPK);
    __bf16* h1     = (__bf16*)(ws + 3 * WPK + NBH * sizeof(__bf16));
    float*  c0     = (float*) (ws + 3 * WPK + 2 * NBH * sizeof(__bf16));
    float*  c1     = (float*) (ws + 3 * WPK + 2 * NBH * sizeof(__bf16) + NBH * sizeof(float));

    // 1) pack weights to bf16 WMMA-fragment layout (L2-resident afterwards)
    int packThreads = 256 * KTILES * 32;          // (n_tiles * k_tiles * lanes)
    pack_weight_kernel<<<packThreads / 256, 256, 0, stream>>>(Whh0, pw_hh0);
    pack_weight_kernel<<<packThreads / 256, 256, 0, stream>>>(Wih1, pw_ih1);
    pack_weight_kernel<<<packThreads / 256, 256, 0, stream>>>(Whh1, pw_hh1);

    // 2) init working state
    init_state_kernel<<<(int)(NBH / 256), 256, 0, stream>>>(eh, ec, h0, h1, c0, c1, (int)NBH);

    // 3) time loop: stream-ordered kernels carry the sequential dependency
    dim3 grid(BDIM / BM, HDIM / 128);
    for (int t = 0; t < TDIM; ++t) {
        const float* xp = (t == 0) ? x0 : (outF + (size_t)(t - 1) * FDIM);
        int xs          = (t == 0) ? FDIM : TDIM * FDIM;
        lstm_layer_kernel<false, true><<<grid, 256, LDS_BYTES, stream>>>(
            h0, pw_hh0, nullptr, nullptr, h0, c0, bih0, bhh0, xp, xs, Wih0);
        lstm_layer_kernel<true, false><<<grid, 256, 2 * LDS_BYTES, stream>>>(
            h0, pw_ih1, h1, pw_hh1, h1, c1, bih1, bhh1, nullptr, 0, nullptr);
        fc_out_kernel<<<BDIM / 4, 128, 0, stream>>>(h1, Wfc, bfc, outF, t);
    }
}
